// _RPN_35811437314316
// MI455X (gfx1250) — matrix operand order repaired and checked by reading, not compiled, not run on hardware
//
#include <hip/hip_runtime.h>
#include <hip/hip_bf16.h>

typedef _Float16 f16;
typedef __attribute__((ext_vector_type(16))) _Float16 v16h;
typedef __attribute__((ext_vector_type(8)))  _Float16 v8h;
typedef __attribute__((ext_vector_type(8)))  float    v8f;
typedef __attribute__((ext_vector_type(4)))  int      v4i;

#define NEGF (-1.0e10f)

#if defined(__has_builtin)
#  if __has_builtin(__builtin_amdgcn_global_load_async_to_lds_b128)
#    define HAVE_ASYNC_LDS 1
#  endif
#endif

#if defined(HAVE_ASYNC_LDS)
typedef __attribute__((address_space(1))) v4i g_v4i;   // global
typedef __attribute__((address_space(3))) v4i l_v4i;   // LDS
__device__ __forceinline__ void async_wait0() {
#if __has_builtin(__builtin_amdgcn_s_wait_asynccnt)
  __builtin_amdgcn_s_wait_asynccnt(0);
#else
  asm volatile("s_wait_asynccnt 0" ::: "memory");
#endif
}
#endif

__device__ __forceinline__ v16h cat8(v8h lo, v8h hi) {
  return __builtin_shufflevector(lo, hi, 0,1,2,3,4,5,6,7,8,9,10,11,12,13,14,15);
}

__device__ __forceinline__ v8f wmma_f16(v16h a, v16h b, v8f c) {
  return __builtin_amdgcn_wmma_f32_16x16x32_f16(false, a, false, b, (short)0, c, false, false);
}

// ---------------- pack kernels ----------------

// base_feat NCHW f32 -> In16 padded NHWC f16 : In16[b][hp 0..65][wp 0..65][c],
// halo ring (hp/wp == 0 or 65) is zero => conv needs no border predicates.
__global__ void pack_input(const float* __restrict__ in, f16* __restrict__ out, int n) {
  int idx = blockIdx.x * blockDim.x + threadIdx.x;
  if (idx >= n) return;
  int c     = idx & 1023;
  int rest  = idx >> 10;          // b*4356 + hp*66 + wp
  int wp    = rest % 66;
  int rest2 = rest / 66;
  int hp    = rest2 % 66;
  int b     = rest2 / 66;
  float v = 0.f;
  if (hp >= 1 && hp <= 64 && wp >= 1 && wp <= 64)
    v = in[((size_t)b * 1024 + c) * 4096 + (size_t)(hp - 1) * 64 + (wp - 1)];
  out[idx] = (f16)v;
}

// conv_w [O=512][I=1024][3][3] f32 -> Wt[t][o][i] f16
__global__ void pack_convw(const float* __restrict__ in, f16* __restrict__ out, int n) {
  int idx = blockIdx.x * blockDim.x + threadIdx.x;
  if (idx >= n) return;
  int i = idx & 1023;
  int o = (idx >> 10) & 511;
  int t = idx >> 19;
  out[idx] = (f16)in[((size_t)o * 1024 + i) * 9 + t];
}

// cls_w [18][512] + bbox_w [36][512] -> Wh[n=64][c=512] f16, rows 54..63 zero
__global__ void pack_headw(const float* __restrict__ clsw, const float* __restrict__ bbxw,
                           f16* __restrict__ out, int n) {
  int idx = blockIdx.x * blockDim.x + threadIdx.x;
  if (idx >= n) return;
  int c = idx & 511;
  int r = idx >> 9;
  float v = 0.f;
  if (r < 18)      v = clsw[(size_t)r * 512 + c];
  else if (r < 54) v = bbxw[(size_t)(r - 18) * 512 + c];
  out[idx] = (f16)v;
}

// ---------------- conv3x3 implicit-GEMM WMMA ----------------
// grid.x = B*H*(W/16) = 1024  (b = bx>>8, h = (bx>>2)&63, w0 = (bx&3)*16)
// block = 256 (8 waves); wave w owns couts [w*64, w*64+64) as 4 WMMA tiles.
// K staged in LDS 128 channels at a time: 1 barrier pair per 16 WMMAs/wave.
__global__ __launch_bounds__(256) void conv3x3_wmma(
    const f16* __restrict__ In16, const f16* __restrict__ Wt,
    const float* __restrict__ conv_b, f16* __restrict__ X16) {
  __shared__ f16 a_lds[16 * 128];   // 16 rows (M) x 128 channels (K), 4 KB

  const unsigned tid  = threadIdx.x;
  const unsigned bx   = blockIdx.x;
  const unsigned b    = bx >> 8;
  const unsigned h    = (bx >> 2) & 63;
  const unsigned w0   = (bx & 3) << 4;
  const unsigned wave = tid >> 5;
  const unsigned lane = tid & 31;
  const unsigned n0   = wave * 64u;

  // cooperative fill: thread -> (row, 16-byte segment of the 256B row)
  const unsigned frow = tid >> 4;   // 0..15
  const unsigned fseg = tid & 15;   // 0..15

  // fragment geometry (contiguous-K 16-bit layout used for A and B, wave32)
  const unsigned nl  = lane & 15;
  const unsigned kbA = (lane < 16) ? 0u : 16u;
  const unsigned kbB = (lane < 16) ? 0u : 16u;
  const unsigned mb  = (lane < 16) ? 0u : 8u;

  v8f acc[4] = {v8f{}, v8f{}, v8f{}, v8f{}};

  #pragma unroll
  for (int t = 0; t < 9; ++t) {
    const int dh = t / 3, dw = t % 3;   // offsets into the padded frame
    const f16* srcRow =
        In16 + (((size_t)b * 66 + (h + dh)) * 66 + (w0 + frow + dw)) * 1024;
    const f16* wtap = Wt + (size_t)t * 512 * 1024;

    for (int fill = 0; fill < 8; ++fill) {   // K chunks of 128 channels
      __syncthreads();
#if defined(HAVE_ASYNC_LDS)
      __builtin_amdgcn_global_load_async_to_lds_b128(
          (g_v4i*)(srcRow + fill * 128 + fseg * 8),
          (l_v4i*)(a_lds + frow * 128 + fseg * 8), 0, 0);
      async_wait0();
#else
      v8h tmp = *(const v8h*)(srcRow + fill * 128 + fseg * 8);
      *(v8h*)(a_lds + frow * 128 + fseg * 8) = tmp;
#endif
      if (fill + 1 < 8) __builtin_prefetch(srcRow + (fill + 1) * 128 + fseg * 8, 0, 1);
      __syncthreads();

      #pragma unroll
      for (int kk = 0; kk < 4; ++kk) {
        // A fragment from LDS: row = lane&15, K = {kbA..+15} contiguous of this 32-chunk
        const f16* ap = a_lds + (size_t)nl * 128 + kk * 32 + kbA;
        v16h af = cat8(*(const v8h*)ap, *(const v8h*)(ap + 8));
        // B fragments: col = lane&15, K contiguous; 4 cout tiles 16 apart
        const f16* bp = wtap + (size_t)(n0 + nl) * 1024 + fill * 128 + kk * 32 + kbB;
        acc[0] = wmma_f16(af, *(const v16h*)(bp),             acc[0]);
        acc[1] = wmma_f16(af, *(const v16h*)(bp + 16 * 1024), acc[1]);
        acc[2] = wmma_f16(af, *(const v16h*)(bp + 32 * 1024), acc[2]);
        acc[3] = wmma_f16(af, *(const v16h*)(bp + 48 * 1024), acc[3]);
      }
    }
  }

  // epilogue: bias + relu, store f16 NHWC [b][pos][cout]
  const size_t posb = (size_t)b * 4096 + h * 64 + w0;
  #pragma unroll
  for (int j = 0; j < 4; ++j) {
    const unsigned n = n0 + j * 16 + nl;
    const float bias = conv_b[n];
    #pragma unroll
    for (int v = 0; v < 8; ++v) {
      float r = acc[j][v] + bias;
      r = r > 0.f ? r : 0.f;
      X16[(posb + mb + v) * 512 + n] = (f16)r;
    }
  }
}

// ---------------- 1x1 head GEMM (cls 18 + bbox 36, padded to N=64) ----------------
// grid.x = 16384/128 = 128, block = 256 (8 waves); wave -> one 16-row M tile,
// A fragment reused across the 4 cout tiles.
__global__ __launch_bounds__(256) void head_wmma(
    const f16* __restrict__ X16, const f16* __restrict__ Wh, float* __restrict__ HO) {
  const unsigned tid  = threadIdx.x;
  const unsigned lane = tid & 31;
  const unsigned wave = tid >> 5;
  const unsigned m0   = blockIdx.x * 128u + wave * 16u;

  const unsigned nl  = lane & 15;
  const unsigned kbA = (lane < 16) ? 0u : 16u;
  const unsigned kbB = (lane < 16) ? 0u : 16u;
  const unsigned mb  = (lane < 16) ? 0u : 8u;

  v8f acc[4] = {v8f{}, v8f{}, v8f{}, v8f{}};
  for (int kc = 0; kc < 16; ++kc) {
    const f16* ap = X16 + (size_t)(m0 + nl) * 512 + kc * 32 + kbA;
    v16h af = cat8(*(const v8h*)ap, *(const v8h*)(ap + 8));
    #pragma unroll
    for (int j = 0; j < 4; ++j) {
      v16h bf = *(const v16h*)(Wh + (size_t)(j * 16 + nl) * 512 + kc * 32 + kbB);
      acc[j] = wmma_f16(af, bf, acc[j]);
    }
  }
  #pragma unroll
  for (int j = 0; j < 4; ++j)
    #pragma unroll
    for (int v = 0; v < 8; ++v)
      HO[(size_t)(m0 + mb + v) * 64 + j * 16 + nl] = acc[j][v];
}

// ---------------- softmax + anchor decode ----------------
__global__ void decode_kernel(const float* __restrict__ HO,
                              const float* __restrict__ cls_b,
                              const float* __restrict__ bbox_b,
                              const float* __restrict__ im_info,
                              float* __restrict__ scores, float* __restrict__ boxes, int n) {
  int idx = blockIdx.x * blockDim.x + threadIdx.x;
  if (idx >= n) return;
  const int b   = idx / 36864;
  const int r   = idx - b * 36864;
  const int a   = r % 9;
  const int pos = r / 9;
  const int hh  = pos >> 6;
  const int ww  = pos & 63;

  const float* hr = HO + ((size_t)b * 4096 + pos) * 64;
  const float s0 = hr[a]     + cls_b[a];
  const float s1 = hr[9 + a] + cls_b[9 + a];
  const float fg = 1.0f / (1.0f + __expf(s0 - s1));

  const float d0 = hr[18 + a * 4 + 0] + bbox_b[a * 4 + 0];
  const float d1 = hr[18 + a * 4 + 1] + bbox_b[a * 4 + 1];
  const float d2 = hr[18 + a * 4 + 2] + bbox_b[a * 4 + 2];
  const float d3 = hr[18 + a * 4 + 3] + bbox_b[a * 4 + 3];

  const int ridx = a / 3, sidx = a % 3;
  const float ratio = (ridx == 0) ? 0.5f : (ridx == 1 ? 1.0f : 2.0f);
  const float scale = (sidx == 0) ? 8.0f : (sidx == 1 ? 16.0f : 32.0f);
  const float wsr = rintf(sqrtf(256.0f / ratio));
  const float hsr = rintf(wsr * ratio);
  const float ws2 = wsr * scale, hs2 = hsr * scale;
  const float sx = (float)ww * 16.0f, sy = (float)hh * 16.0f;
  const float ax1 = 7.5f - 0.5f * (ws2 - 1.0f) + sx;
  const float ay1 = 7.5f - 0.5f * (hs2 - 1.0f) + sy;
  const float ax2 = 7.5f + 0.5f * (ws2 - 1.0f) + sx;
  const float ay2 = 7.5f + 0.5f * (hs2 - 1.0f) + sy;

  const float wa = ax2 - ax1 + 1.0f, ha = ay2 - ay1 + 1.0f;
  const float cxa = ax1 + 0.5f * (wa - 1.0f), cya = ay1 + 0.5f * (ha - 1.0f);
  const float cx = d0 * wa + cxa, cy = d1 * ha + cya;
  const float pw = __expf(d2) * wa, ph = __expf(d3) * ha;

  const float imh = im_info[b * 3 + 0], imw = im_info[b * 3 + 1];
  float x1 = fminf(fmaxf(cx - 0.5f * pw, 0.0f), imw - 1.0f);
  float y1 = fminf(fmaxf(cy - 0.5f * ph, 0.0f), imh - 1.0f);
  float x2 = fminf(fmaxf(cx + 0.5f * pw, 0.0f), imw - 1.0f);
  float y2 = fminf(fmaxf(cy + 0.5f * ph, 0.0f), imh - 1.0f);

  scores[idx] = fg;
  boxes[(size_t)idx * 4 + 0] = x1;
  boxes[(size_t)idx * 4 + 1] = y1;
  boxes[(size_t)idx * 4 + 2] = x2;
  boxes[(size_t)idx * 4 + 3] = y2;
}

// ---------------- greedy NMS: one block per image ----------------
__global__ __launch_bounds__(1024) void nms_kernel(
    float* __restrict__ scores, const float* __restrict__ boxes,
    float* __restrict__ rois, float* __restrict__ kept) {
  const int b = blockIdx.x;
  float* sc = scores + (size_t)b * 36864;
  const float* bxp = boxes + (size_t)b * 36864 * 4;
  __shared__ float sv[1024];
  __shared__ int   si[1024];
  __shared__ float pick[5];
  const int N = 36864;

  for (int i = 0; i < 300; ++i) {
    float bestv = -3.0e38f;
    int   besti = 0x7fffffff;
    for (int j = threadIdx.x; j < N; j += 1024) {
      const float v = sc[j];
      if (v > bestv || (v == bestv && j < besti)) { bestv = v; besti = j; }
    }
    sv[threadIdx.x] = bestv;
    si[threadIdx.x] = besti;
    __syncthreads();
    for (int s = 512; s > 0; s >>= 1) {
      if ((int)threadIdx.x < s) {
        const float ov = sv[threadIdx.x + s];
        const int   oi = si[threadIdx.x + s];
        if (ov > sv[threadIdx.x] || (ov == sv[threadIdx.x] && oi < si[threadIdx.x])) {
          sv[threadIdx.x] = ov; si[threadIdx.x] = oi;
        }
      }
      __syncthreads();
    }
    if (threadIdx.x == 0) {
      const int   idx = si[0];
      const float s   = sv[0];
      const bool  valid = s > NEGF * 0.5f;
      const float x1 = bxp[(size_t)idx * 4 + 0], y1 = bxp[(size_t)idx * 4 + 1];
      const float x2 = bxp[(size_t)idx * 4 + 2], y2 = bxp[(size_t)idx * 4 + 3];
      pick[0] = x1; pick[1] = y1; pick[2] = x2; pick[3] = y2;
      pick[4] = (x2 - x1 + 1.0f) * (y2 - y1 + 1.0f);
      float* ro = rois + ((size_t)b * 300 + i) * 5;
      ro[0] = (float)b;
      ro[1] = valid ? x1 : 0.0f; ro[2] = valid ? y1 : 0.0f;
      ro[3] = valid ? x2 : 0.0f; ro[4] = valid ? y2 : 0.0f;
      kept[(size_t)b * 300 + i] = valid ? s : 0.0f;
    }
    __syncthreads();
    const float px1 = pick[0], py1 = pick[1], px2 = pick[2], py2 = pick[3], pa = pick[4];
    for (int j = threadIdx.x; j < N; j += 1024) {
      const float x1 = bxp[(size_t)j * 4 + 0], y1 = bxp[(size_t)j * 4 + 1];
      const float x2 = bxp[(size_t)j * 4 + 2], y2 = bxp[(size_t)j * 4 + 3];
      const float xx1 = fmaxf(px1, x1), yy1 = fmaxf(py1, y1);
      const float xx2 = fminf(px2, x2), yy2 = fminf(py2, y2);
      const float inter = fmaxf(xx2 - xx1 + 1.0f, 0.0f) * fmaxf(yy2 - yy1 + 1.0f, 0.0f);
      const float area  = (x2 - x1 + 1.0f) * (y2 - y1 + 1.0f);
      const float iou   = inter / (pa + area - inter);
      if (iou > 0.7f) sc[j] = NEGF;
    }
    __syncthreads();
  }
}

// ---------------- host launcher ----------------
extern "C" void kernel_launch(void* const* d_in, const int* in_sizes, int n_in,
                              void* d_out, int out_size, void* d_ws, size_t ws_size,
                              hipStream_t stream) {
  (void)in_sizes; (void)n_in; (void)out_size; (void)ws_size;
  const float* base_feat = (const float*)d_in[0];
  const float* im_info   = (const float*)d_in[1];
  const float* conv_w    = (const float*)d_in[4];
  const float* conv_b    = (const float*)d_in[5];
  const float* cls_w     = (const float*)d_in[6];
  const float* cls_b     = (const float*)d_in[7];
  const float* bbox_w    = (const float*)d_in[8];
  const float* bbox_b    = (const float*)d_in[9];
  float* out = (float*)d_out;

  char* ws = (char*)d_ws;
  size_t off = 0;
  auto carve = [&](size_t bytes) { size_t r = off; off = (off + bytes + 255) & ~(size_t)255; return r; };
  f16*   In16   = (f16*)  (ws + carve((size_t)4 * 66 * 66 * 1024 * 2));  // 34 MB (padded)
  f16*   Wt     = (f16*)  (ws + carve((size_t)9 * 512 * 1024 * 2));      // 9 MB
  f16*   X16    = (f16*)  (ws + carve((size_t)4 * 4096 * 512 * 2));      // 16 MB
  f16*   Wh     = (f16*)  (ws + carve((size_t)64 * 512 * 2));            // 64 KB
  float* HO     = (float*)(ws + carve((size_t)16384 * 64 * 4));          // 4 MB
  float* scores = (float*)(ws + carve((size_t)4 * 36864 * 4));           // 576 KB
  float* boxes  = (float*)(ws + carve((size_t)4 * 36864 * 4 * 4));       // 2.25 MB

  {
    const int n = 4 * 66 * 66 * 1024;
    pack_input<<<(n + 255) / 256, 256, 0, stream>>>(base_feat, In16, n);
  }
  {
    const int n = 9 * 512 * 1024;
    pack_convw<<<(n + 255) / 256, 256, 0, stream>>>(conv_w, Wt, n);
  }
  {
    const int n = 64 * 512;
    pack_headw<<<(n + 255) / 256, 256, 0, stream>>>(cls_w, bbox_w, Wh, n);
  }

  conv3x3_wmma<<<1024, 256, 0, stream>>>(In16, Wt, conv_b, X16);
  head_wmma<<<128, 256, 0, stream>>>(X16, Wh, HO);

  {
    const int n = 4 * 36864;
    decode_kernel<<<(n + 255) / 256, 256, 0, stream>>>(HO, cls_b, bbox_b, im_info, scores, boxes, n);
  }

  float* rois = out;                 // [4][300][5]
  float* kept = out + 4 * 300 * 5;   // [4][300]
  nms_kernel<<<4, 1024, 0, stream>>>(scores, boxes, rois, kept);
}
